// GNN_25864293056529
// MI455X (gfx1250) — matrix operand order repaired and checked by reading, not compiled, not run on hardware
//
#include <hip/hip_runtime.h>
#include <hip/hip_bf16.h>

#define N_NODES 100000
#define N_EDGES 3200000
#define F_HID   256
#define N_CLS   40

typedef __attribute__((ext_vector_type(16))) _Float16 v16h;
typedef __attribute__((ext_vector_type(8)))  float    v8f;

// No-return f32 atomic add -> global_atomic_add_f32 (L2 atomic unit, STOREcnt-tracked).
__device__ __forceinline__ void atom_add_f32(float* p, float v) {
    asm volatile("global_atomic_add_f32 %0, %1, off" :: "v"(p), "v"(v) : "memory");
}

// ---------------- degree / norm ----------------
__global__ void degree_kernel(const int* __restrict__ src, const int* __restrict__ dst,
                              float* deg_out, float* deg_in, int n_edges) {
    int e = blockIdx.x * blockDim.x + threadIdx.x;
    if (e >= n_edges) return;
    atom_add_f32(deg_out + src[e], 1.0f);
    atom_add_f32(deg_in + dst[e], 1.0f);
}

__global__ void norm_kernel(float* norm_out, float* norm_in, int n) {
    int i = blockIdx.x * blockDim.x + threadIdx.x;
    if (i >= n) return;
    norm_out[i] = rsqrtf(fmaxf(norm_out[i], 1.0f));
    norm_in[i]  = rsqrtf(fmaxf(norm_in[i], 1.0f));
}

// ---------------- edge aggregation (SpMM), 256-wide: one wave32 per edge ----------------
// agg[dst] += h[src] * norm_out[src]; each lane handles 8 consecutive floats (2x float4).
__global__ void scatter256_kernel(const float* __restrict__ h, const int* __restrict__ src,
                                  const int* __restrict__ dst, const float* __restrict__ norm_out,
                                  float* __restrict__ agg, int n_edges) {
    int wid  = (blockIdx.x * blockDim.x + threadIdx.x) >> 5;
    int lane = threadIdx.x & 31;
    if (wid >= n_edges) return;
    int s = src[wid], d = dst[wid];
    float no = norm_out[s];
    const float4* hp = reinterpret_cast<const float4*>(h + (size_t)s * F_HID + lane * 8);
    float4 x0 = hp[0];
    float4 x1 = hp[1];
    float* ap = agg + (size_t)d * F_HID + lane * 8;
    atom_add_f32(ap + 0, x0.x * no);
    atom_add_f32(ap + 1, x0.y * no);
    atom_add_f32(ap + 2, x0.z * no);
    atom_add_f32(ap + 3, x0.w * no);
    atom_add_f32(ap + 4, x1.x * no);
    atom_add_f32(ap + 5, x1.y * no);
    atom_add_f32(ap + 6, x1.z * no);
    atom_add_f32(ap + 7, x1.w * no);
}

// 40-wide variant for the classifier layer (norm_out fused; row-scaling commutes with @W).
__global__ void scatter40_kernel(const float* __restrict__ t, const int* __restrict__ src,
                                 const int* __restrict__ dst, const float* __restrict__ norm_out,
                                 float* __restrict__ agg, int n_edges) {
    int wid  = (blockIdx.x * blockDim.x + threadIdx.x) >> 5;
    int lane = threadIdx.x & 31;
    if (wid >= n_edges) return;
    int s = src[wid], d = dst[wid];
    float no = norm_out[s];
    for (int f = lane; f < N_CLS; f += 32)
        atom_add_f32(agg + (size_t)d * N_CLS + f, t[(size_t)s * N_CLS + f] * no);
}

// ---------------- WMMA GEMM: C[M,N] = A[M,K] @ W[K,N], optional (*norm_in + bias, relu) ----
// One wave per 16x16 output tile. W column-tile staged to LDS as f16 with 264-half pitch
// (16B-aligned per-lane 16-half runs, conflict-free banks).
#define WL_PITCH 264
__global__ void __launch_bounds__(256)
wmma_gemm_kernel(const float* __restrict__ A, const float* __restrict__ W,
                 float* __restrict__ C, const float* __restrict__ norm_in,
                 const float* __restrict__ bias, int M, int N, int K,
                 int do_epilogue, int do_relu) {
    __shared__ _Float16 Wl[16 * WL_PITCH];
    const int n0 = blockIdx.y * 16;

    // Stage W[:, n0:n0+16] -> LDS (f16, transposed: Wl[c*PITCH + k]).
    for (int k = threadIdx.x; k < K; k += blockDim.x) {
        const float* wr = W + (size_t)k * N;
        #pragma unroll
        for (int c = 0; c < 16; ++c) {
            float w = (n0 + c < N) ? wr[n0 + c] : 0.0f;
            Wl[c * WL_PITCH + k] = (_Float16)w;
        }
    }
    __syncthreads();

    const int wave   = threadIdx.x >> 5;
    const int lane   = threadIdx.x & 31;
    const int tile_m = blockIdx.x * (blockDim.x >> 5) + wave;
    if (tile_m * 16 >= M) return;          // wave-uniform exit (after the only barrier)
    const int m0   = tile_m * 16;
    const int hsel = lane >> 4;            // 0: lanes 0-15, 1: lanes 16-31
    const int lid  = lane & 15;

    v8f acc = {};
    const float*    Arow = A + (size_t)(m0 + lid) * K;   // A row for this lane (ISA layout)
    const _Float16* Bcol = &Wl[lid * WL_PITCH];          // B column for this lane

    for (int kb = 0; kb < K; kb += 32) {
        v16h a, bm;
        // A, contiguous k order used here (same as B): lanes 0-15 hold K kb+0..15, lanes 16-31 hold K kb+16..31.
        const float* ap = Arow + kb + hsel * 16;
        #pragma unroll
        for (int e = 0; e < 8; ++e)  a[e] = (_Float16)ap[e];
        #pragma unroll
        for (int e = 8; e < 16; ++e) a[e] = (_Float16)ap[e];
        // B, contiguous k order used here (same as A): lanes 0-15 hold K kb+0..15, lanes 16-31 hold K kb+16..31.
        const _Float16* bp = Bcol + kb + hsel * 16;
        #pragma unroll
        for (int e = 0; e < 16; ++e) bm[e] = bp[e];

        acc = __builtin_amdgcn_wmma_f32_16x16x32_f16(false, a, false, bm,
                                                     (short)0, acc, false, false);
    }

    // C/D layout: VGPR r -> row m0 + hsel*8 + r, col n0 + lid.
    const int col = n0 + lid;
    if (col < N) {
        float bb = do_epilogue ? bias[col] : 0.0f;
        #pragma unroll
        for (int r = 0; r < 8; ++r) {
            int row = m0 + hsel * 8 + r;
            float v = acc[r];
            if (do_epilogue) {
                v = v * norm_in[row] + bb;
                if (do_relu) v = fmaxf(v, 0.0f);
            }
            C[(size_t)row * N + col] = v;
        }
    }
}

// ---------------- final epilogue for classifier layer ----------------
__global__ void epilogue40_kernel(const float* __restrict__ agg, const float* __restrict__ norm_in,
                                  const float* __restrict__ bias, float* __restrict__ out,
                                  int n_nodes) {
    int idx = blockIdx.x * blockDim.x + threadIdx.x;
    if (idx >= n_nodes * N_CLS) return;
    int i = idx / N_CLS, c = idx % N_CLS;
    out[idx] = agg[idx] * norm_in[i] + bias[c];
}

extern "C" void kernel_launch(void* const* d_in, const int* in_sizes, int n_in,
                              void* d_out, int out_size, void* d_ws, size_t ws_size,
                              hipStream_t stream) {
    const float* features = (const float*)d_in[0];
    const int*   src      = (const int*)d_in[1];
    const int*   dst      = (const int*)d_in[2];
    const float* W0 = (const float*)d_in[3];
    const float* b0 = (const float*)d_in[4];
    const float* W1 = (const float*)d_in[5];
    const float* b1 = (const float*)d_in[6];
    const float* W2 = (const float*)d_in[7];
    const float* b2 = (const float*)d_in[8];
    float* out = (float*)d_out;

    // Workspace layout (floats): norms (2*100K) | agg (100K*256) | hbuf (100K*256)
    float* ws       = (float*)d_ws;
    float* norm_out = ws;
    float* norm_in  = norm_out + N_NODES;
    float* agg      = norm_in + N_NODES;
    float* hbuf     = agg + (size_t)N_NODES * F_HID;
    float* t2       = agg;                              // reuse after layer-1 GEMM consumed agg
    float* agg2     = agg + (size_t)N_NODES * N_CLS;

    const int B = 256;
    const unsigned edge_waves_blocks = (unsigned)(((size_t)N_EDGES * 32 + B - 1) / B);
    dim3 ggrid((6250 + 7) / 8, F_HID / 16);             // 100000/16 = 6250 M-tiles
    dim3 ggrid2((6250 + 7) / 8, (N_CLS + 15) / 16);

    // Degrees (accumulated in-place into norm buffers) -> norms.
    hipMemsetAsync(norm_out, 0, 2 * (size_t)N_NODES * sizeof(float), stream);
    degree_kernel<<<(N_EDGES + B - 1) / B, B, 0, stream>>>(src, dst, norm_out, norm_in, N_EDGES);
    norm_kernel<<<(N_NODES + B - 1) / B, B, 0, stream>>>(norm_out, norm_in, N_NODES);

    // Layer 0: agg = A_norm @ (features*norm_out); h1 = relu(agg@W0 * norm_in + b0)
    hipMemsetAsync(agg, 0, (size_t)N_NODES * F_HID * sizeof(float), stream);
    scatter256_kernel<<<edge_waves_blocks, B, 0, stream>>>(features, src, dst, norm_out, agg, N_EDGES);
    wmma_gemm_kernel<<<ggrid, B, 0, stream>>>(agg, W0, hbuf, norm_in, b0,
                                              N_NODES, F_HID, F_HID, 1, 1);
    // Layer 1
    hipMemsetAsync(agg, 0, (size_t)N_NODES * F_HID * sizeof(float), stream);
    scatter256_kernel<<<edge_waves_blocks, B, 0, stream>>>(hbuf, src, dst, norm_out, agg, N_EDGES);
    wmma_gemm_kernel<<<ggrid, B, 0, stream>>>(agg, W1, hbuf, norm_in, b1,
                                              N_NODES, F_HID, F_HID, 1, 1);
    // Layer 2: t = h2@W2 (plain), then aggregate 40-wide, then *norm_in + b2.
    wmma_gemm_kernel<<<ggrid2, B, 0, stream>>>(hbuf, W2, t2, nullptr, nullptr,
                                               N_NODES, N_CLS, F_HID, 0, 0);
    hipMemsetAsync(agg2, 0, (size_t)N_NODES * N_CLS * sizeof(float), stream);
    scatter40_kernel<<<edge_waves_blocks, B, 0, stream>>>(t2, src, dst, norm_out, agg2, N_EDGES);
    epilogue40_kernel<<<((unsigned)(N_NODES * N_CLS) + B - 1) / B, B, 0, stream>>>(
        agg2, norm_in, b2, out, N_NODES);
}
